// FusionGNN_41601053229622
// MI455X (gfx1250) — compile-verified, matrix operand order checked by reading, not run on hardware
//
#include <hip/hip_runtime.h>

typedef __attribute__((ext_vector_type(16))) _Float16 v16h;
typedef __attribute__((ext_vector_type(8)))  _Float16 v8h;
typedef __attribute__((ext_vector_type(8)))  float    v8f;

#define KC   256           // K-chunk staged in LDS per block iteration
#define KCP  (KC + 8)      // padded column stride in halves (16B-aligned: 528B)

// ---------------------------------------------------------------------------
// Elementwise conversion / graph kernels
// ---------------------------------------------------------------------------
__global__ void k_zero(float* __restrict__ p, int n) {
    int i = blockIdx.x * blockDim.x + threadIdx.x;
    if (i < n) p[i] = 0.0f;
}

__global__ void k_count_deg(const int* __restrict__ col, float* __restrict__ deg, int E) {
    int i = blockIdx.x * blockDim.x + threadIdx.x;
    if (i < E) atomicAdd(&deg[col[i]], 1.0f);
}

__global__ void k_finalize_dis(float* __restrict__ deg, int n) {
    int i = blockIdx.x * blockDim.x + threadIdx.x;
    if (i < n) deg[i] = rsqrtf(deg[i] + 1.0f);   // deg includes +1 self loop
}

// generic f32 -> f16
__global__ void k_cvt(const float* __restrict__ in, _Float16* __restrict__ out, long long n) {
    long long i = (long long)blockIdx.x * blockDim.x + threadIdx.x;
    if (i < n) out[i] = (_Float16)in[i];
}

// xAh[m][0..din-1] = in1[m], xAh[m][din..2din-1] = in2[m], as f16
__global__ void k_concat_cvt(const float* __restrict__ a, const float* __restrict__ b,
                             _Float16* __restrict__ out, int M, int din) {
    long long i = (long long)blockIdx.x * blockDim.x + threadIdx.x;
    long long total = (long long)M * (2 * din);
    if (i < total) {
        int m = (int)(i / (2 * din));
        int c = (int)(i - (long long)m * (2 * din));
        float v = (c < din) ? a[(size_t)m * din + c] : b[(size_t)m * din + (c - din)];
        out[i] = (_Float16)v;
    }
}

// fused relu + f32->f16 (activation feeding the next GEMM)
__global__ void k_relu_cvt(const float* __restrict__ in, _Float16* __restrict__ out, long long n) {
    long long i = (long long)blockIdx.x * blockDim.x + threadIdx.x;
    if (i < n) {
        float v = in[i];
        out[i] = (_Float16)(v > 0.0f ? v : 0.0f);
    }
}

// out[m][c] = h[m][c] * dis[m]^2 + bias[c]   (self-loop contribution + bias)
__global__ void k_init_self(const float* __restrict__ h, const float* __restrict__ dis,
                            const float* __restrict__ bias, float* __restrict__ out,
                            int M, int hid) {
    long long i = (long long)blockIdx.x * blockDim.x + threadIdx.x;
    long long total = (long long)M * hid;
    if (i < total) {
        int m = (int)(i / hid);
        int c = (int)(i - (long long)m * hid);
        float d = dis[m];
        out[i] = h[i] * d * d + bias[c];
    }
}

// one wave per edge: out[col] += h[row] * dis[row]*dis[col]
__global__ void __launch_bounds__(256) k_scatter(const float* __restrict__ h,
                                                 const int* __restrict__ row,
                                                 const int* __restrict__ col,
                                                 const float* __restrict__ dis,
                                                 float* __restrict__ out, int E, int hid) {
    int e = blockIdx.x * (blockDim.x >> 5) + (threadIdx.x >> 5);
    int lane = threadIdx.x & 31;
    if (e >= E) return;
    int r = row[e];
    int c = col[e];
    float norm = dis[r] * dis[c];
    const float* hr = h + (size_t)r * hid;
    float* oc = out + (size_t)c * hid;
    for (int i = lane; i < hid; i += 32)
        atomicAdd(&oc[i], hr[i] * norm);
}

// ---------------------------------------------------------------------------
// WMMA fragment helpers (ISA 16-bit A 16x32 / B 32x16 per-lane layouts)
// ---------------------------------------------------------------------------
__device__ inline v16h make_frag(v8h lo, v8h hi16) {
    v16h f;
#pragma unroll
    for (int i = 0; i < 8; ++i) { f[i] = lo[i]; f[8 + i] = hi16[i]; }
    return f;
}

__device__ inline v16h load_b_frag(const _Float16* __restrict__ bp) {
    v16h b;
#pragma unroll
    for (int i = 0; i < 8; ++i) { b[i] = bp[i]; b[8 + i] = bp[16 + i]; }
    return b;
}

// ---------------------------------------------------------------------------
// WMMA GEMM (all-f16 operands, f32 accumulate):
//   out[M,N] = Ah[M,K] * Wh[K,N] (+bias)
// Block = 256 threads = 8 waves, covering 256 rows x 64 cols.
// Each wave owns TWO 16-row stripes and FOUR 16-column tiles (8 accumulators).
// Per K-step, ALL operand fragments (2x A from global, 4x B from LDS) are
// loaded into distinct register tuples before the 8 v_wmma issue, so the
// compiler can use partial dscnt/loadcnt waits and stream loads under the
// WMMA chain. Weights stream through LDS in 256-K chunks (33KB padded).
// ---------------------------------------------------------------------------
__global__ void __launch_bounds__(256) k_gemm_wmma(
    const _Float16* __restrict__ Ah, int K,
    const _Float16* __restrict__ Wh, const float* __restrict__ bias,
    float* __restrict__ out, int M, int N)
{
    __shared__ _Float16 Bsh[64 * KCP];

    const int tid   = threadIdx.x;
    const int ncol0 = blockIdx.y * 64;
    const int lane  = tid & 31;
    const int wave  = tid >> 5;
    const int n     = lane & 15;     // C column-in-tile / B column / A row
    const int hi    = lane >> 4;     // which K-half this lane owns
    const int kb    = hi << 3;       // 0 or 8
    const int tileM = blockIdx.x * 256 + wave * 32;   // 32 rows per wave

    size_t mA0 = (size_t)tileM + n;
    size_t mA1 = (size_t)tileM + 16 + n;
    if (mA0 >= (size_t)M) mA0 = (size_t)M - 1;        // clamp; stores guarded
    if (mA1 >= (size_t)M) mA1 = (size_t)M - 1;
    const _Float16* Arow0 = Ah + mA0 * (size_t)K;
    const _Float16* Arow1 = Ah + mA1 * (size_t)K;

    v8f acc00 = {}, acc01 = {}, acc02 = {}, acc03 = {};   // rows 0..15
    v8f acc10 = {}, acc11 = {}, acc12 = {}, acc13 = {};   // rows 16..31

    for (int kc = 0; kc < K; kc += KC) {
        __syncthreads();   // previous chunk's reads done
        // Stage Wh[kc..kc+KC, ncol0..ncol0+63] -> Bsh[col][k] (f16, padded).
        // Each thread reads a contiguous v8h of 8 columns of one K-row
        // (8 lanes = one full 128B row) and scatters 8 ds_store_b16.
#pragma unroll
        for (int it = 0; it < (KC * 64) / (8 * 256); ++it) {
            int v  = it * 256 + tid;
            int j8 = v & 7;          // which group of 8 columns
            int k  = v >> 3;         // K-row within chunk
            v8h w = *(const v8h*)(Wh + (size_t)(kc + k) * N + ncol0 + j8 * 8);
#pragma unroll
            for (int i = 0; i < 8; ++i)
                Bsh[(j8 * 8 + i) * KCP + k] = w[i];
        }
        __syncthreads();

        if (kc + KC < K) {           // prefetch next A chunk for this lane
            __builtin_prefetch(Arow0 + kc + KC + kb, 0, 1);
            __builtin_prefetch(Arow1 + kc + KC + kb, 0, 1);
        }

#pragma unroll
        for (int k0 = 0; k0 < KC; k0 += 32) {
            const int cg = kc + k0 + kb;           // global K column base
            const int kl = k0 + kb;                // K col base within chunk

            // ---- issue ALL loads first (distinct destination tuples) ----
            v8h a0lo = *(const v8h*)(Arow0 + cg);
            v8h a0hi = *(const v8h*)(Arow0 + cg + 16);
            v8h a1lo = *(const v8h*)(Arow1 + cg);
            v8h a1hi = *(const v8h*)(Arow1 + cg + 16);
            v16h b0 = load_b_frag(&Bsh[( 0 + n) * KCP + kl]);
            v16h b1 = load_b_frag(&Bsh[(16 + n) * KCP + kl]);
            v16h b2 = load_b_frag(&Bsh[(32 + n) * KCP + kl]);
            v16h b3 = load_b_frag(&Bsh[(48 + n) * KCP + kl]);
            v16h a0 = make_frag(a0lo, a0hi);
            v16h a1 = make_frag(a1lo, a1hi);

            // ---- 8 WMMAs in B-arrival order ----
            acc00 = __builtin_amdgcn_wmma_f32_16x16x32_f16(false, a0, false, b0, (short)0, acc00, false, false);
            acc10 = __builtin_amdgcn_wmma_f32_16x16x32_f16(false, a1, false, b0, (short)0, acc10, false, false);
            acc01 = __builtin_amdgcn_wmma_f32_16x16x32_f16(false, a0, false, b1, (short)0, acc01, false, false);
            acc11 = __builtin_amdgcn_wmma_f32_16x16x32_f16(false, a1, false, b1, (short)0, acc11, false, false);
            acc02 = __builtin_amdgcn_wmma_f32_16x16x32_f16(false, a0, false, b2, (short)0, acc02, false, false);
            acc12 = __builtin_amdgcn_wmma_f32_16x16x32_f16(false, a1, false, b2, (short)0, acc12, false, false);
            acc03 = __builtin_amdgcn_wmma_f32_16x16x32_f16(false, a0, false, b3, (short)0, acc03, false, false);
            acc13 = __builtin_amdgcn_wmma_f32_16x16x32_f16(false, a1, false, b3, (short)0, acc13, false, false);
        }
    }

    float bv0 = bias ? bias[ncol0 + n +  0] : 0.0f;
    float bv1 = bias ? bias[ncol0 + n + 16] : 0.0f;
    float bv2 = bias ? bias[ncol0 + n + 32] : 0.0f;
    float bv3 = bias ? bias[ncol0 + n + 48] : 0.0f;

#pragma unroll
    for (int v = 0; v < 8; ++v) {
        int m0 = tileM + v + hi * 8;        // ISA C layout: lanes 16-31 -> +8
        if (m0 < M) {
            float* op = out + (size_t)m0 * N + ncol0 + n;
            op[ 0] = acc00[v] + bv0;
            op[16] = acc01[v] + bv1;
            op[32] = acc02[v] + bv2;
            op[48] = acc03[v] + bv3;
        }
        int m1 = tileM + 16 + v + hi * 8;
        if (m1 < M) {
            float* op = out + (size_t)m1 * N + ncol0 + n;
            op[ 0] = acc10[v] + bv0;
            op[16] = acc11[v] + bv1;
            op[32] = acc12[v] + bv2;
            op[48] = acc13[v] + bv3;
        }
    }
}

// ---------------------------------------------------------------------------
// Launch
// ---------------------------------------------------------------------------
extern "C" void kernel_launch(void* const* d_in, const int* in_sizes, int n_in,
                              void* d_out, int out_size, void* d_ws, size_t ws_size,
                              hipStream_t stream) {
    const float* in1 = (const float*)d_in[0];
    const float* in2 = (const float*)d_in[1];
    const int*   eix = (const int*)  d_in[2];
    const float* W1  = (const float*)d_in[3];
    const float* b1  = (const float*)d_in[4];
    const float* W2  = (const float*)d_in[5];
    const float* b2  = (const float*)d_in[6];
    const float* Wfc = (const float*)d_in[7];
    const float* bfc = (const float*)d_in[8];

    const int hid     = in_sizes[4];            // 256
    const int two_din = in_sizes[3] / hid;      // 1024
    const int din     = two_din / 2;            // 512
    const int M       = in_sizes[0] / din;      // 50000
    const int dout    = in_sizes[8];            // 512
    const int E       = in_sizes[2] / 2;        // 400000

    const int* row = eix;
    const int* col = eix + E;

    // Workspace: dis | h | x | xAh (f16 concat; reused as relu-f16 buf) | W*h
    char* ws = (char*)d_ws;
    size_t o = 0;
    float*    dis = (float*)(ws + o);    o += (((size_t)M * 4)             + 255) & ~(size_t)255;
    float*    h   = (float*)(ws + o);    o += (((size_t)M * hid * 4)       + 255) & ~(size_t)255;
    float*    x   = (float*)(ws + o);    o += (((size_t)M * hid * 4)       + 255) & ~(size_t)255;
    _Float16* xAh = (_Float16*)(ws + o); o += (((size_t)M * two_din * 2)   + 255) & ~(size_t)255;
    _Float16* W1h = (_Float16*)(ws + o); o += (((size_t)two_din * hid * 2) + 255) & ~(size_t)255;
    _Float16* W2h = (_Float16*)(ws + o); o += (((size_t)hid * hid * 2)     + 255) & ~(size_t)255;
    _Float16* Wfh = (_Float16*)(ws + o);
    _Float16* xh  = xAh;                 // aliased: xAh dead after GEMM1
    float* outp = (float*)d_out;

    const int T = 256;
    long long nW1 = (long long)two_din * hid, nW2 = (long long)hid * hid,
              nWf = (long long)hid * dout;
    long long nXA = (long long)M * two_din, nX = (long long)M * hid;

    // --- weight conversion (tiny, once per call) ---
    k_cvt<<<(unsigned)((nW1 + T - 1) / T), T, 0, stream>>>(W1,  W1h, nW1);
    k_cvt<<<(unsigned)((nW2 + T - 1) / T), T, 0, stream>>>(W2,  W2h, nW2);
    k_cvt<<<(unsigned)((nWf + T - 1) / T), T, 0, stream>>>(Wfc, Wfh, nWf);

    // --- degree / normalization ---
    k_zero        <<<(M + T - 1) / T, T, 0, stream>>>(dis, M);
    k_count_deg   <<<(E + T - 1) / T, T, 0, stream>>>(col, dis, E);
    k_finalize_dis<<<(M + T - 1) / T, T, 0, stream>>>(dis, M);

    // --- A-side f16 prep: xAh = (f16) concat(in1, in2) ---
    k_concat_cvt<<<(unsigned)((nXA + T - 1) / T), T, 0, stream>>>(in1, in2, xAh, M, din);

    // --- layer 1: h = xAh @ W1 ---
    dim3 g1((M + 255) / 256, hid / 64);
    k_gemm_wmma<<<g1, T, 0, stream>>>(xAh, two_din, W1h, nullptr, h, M, hid);
    k_init_self<<<(unsigned)((nX + T - 1) / T), T, 0, stream>>>(h, dis, b1, x, M, hid);
    k_scatter  <<<(E + 7) / 8, T, 0, stream>>>(h, row, col, dis, x, E, hid);

    // --- layer 2: h = relu(x) @ W2 ---
    k_relu_cvt<<<(unsigned)((nX + T - 1) / T), T, 0, stream>>>(x, xh, nX);
    k_gemm_wmma<<<g1, T, 0, stream>>>(xh, hid, W2h, nullptr, h, M, hid);
    k_init_self<<<(unsigned)((nX + T - 1) / T), T, 0, stream>>>(h, dis, b2, x, M, hid);
    k_scatter  <<<(E + 7) / 8, T, 0, stream>>>(h, row, col, dis, x, E, hid);

    // --- FC head: out = relu(x) @ Wfc + bfc ---
    k_relu_cvt<<<(unsigned)((nX + T - 1) / T), T, 0, stream>>>(x, xh, nX);
    dim3 g3((M + 255) / 256, dout / 64);
    k_gemm_wmma<<<g3, T, 0, stream>>>(xh, hid, Wfh, bfc, outp, M, dout);
}
